// LightweightPatchAttention_39152921870505
// MI455X (gfx1250) — matrix operand order repaired and checked by reading, not compiled, not run on hardware
//
#include <hip/hip_runtime.h>
#include <hip/hip_bf16.h>
#include <stdint.h>

// Shapes fixed by the harness setup
#define B_      8
#define C_      256
#define H_      256
#define W_      256
#define PATCH_  8
#define NH_     (H_/PATCH_)          // 32
#define NW_     (W_/PATCH_)          // 32
#define NPATCH_ (B_*NH_*NW_)         // 8192
#define CR_     (C_/8)               // 32 (C // r)
#define EPS_    1e-5f

typedef __attribute__((ext_vector_type(16))) __bf16 v16bf;
typedef __attribute__((ext_vector_type(8)))  float  v8f;
typedef __attribute__((ext_vector_type(4)))  float  v4f;   // clang vector: OK for nontemporal builtins

__device__ __forceinline__ unsigned short f2bf(float f) {
  unsigned u = __float_as_uint(f);
  u += 0x7FFFu + ((u >> 16) & 1u);      // round-to-nearest-even
  return (unsigned short)(u >> 16);
}
// fast sigmoid / silu: v_exp_f32 + v_rcp_f32 (no IEEE div chains)
__device__ __forceinline__ float sigm_(float x) {
  return __builtin_amdgcn_rcpf(1.0f + __expf(-x));
}
__device__ __forceinline__ float silu_(float x) { return x * sigm_(x); }

// ---------------------------------------------------------------------------
// K1: one workgroup per 8x8 patch. Async-stage patch (256 ch x 64 px, 64 KiB)
// into LDS, compute per-pixel LN stats and per-patch per-channel means (pm).
// ---------------------------------------------------------------------------
__global__ __launch_bounds__(256) void k1_patch_stats(
    const float* __restrict__ x, const float* __restrict__ gamma,
    const float* __restrict__ beta, float* __restrict__ mu_g,
    float* __restrict__ rstd_g, unsigned short* __restrict__ pm_g)
{
  __shared__ __align__(16) float sx[C_ * 64];   // [c][pixel], 64 KiB
  __shared__ float s_sum[4][64];
  __shared__ float s_sq[4][64];
  __shared__ float s_mu[64];
  __shared__ float s_rstd[64];
  __shared__ float s_S;

  const int tid = threadIdx.x;
  const int pidx = blockIdx.x;
  const int b  = pidx >> 10;
  const int pi = (pidx >> 5) & 31;
  const int pj = pidx & 31;
  const int h0 = pi * PATCH_;
  const int w0 = pj * PATCH_;

  // ---- async stage: 4096 x 16B chunks, 16 per thread (CDNA5 async-to-LDS) ----
  for (int q = tid; q < C_ * 16; q += 256) {
    const int c    = q >> 4;
    const int rr   = (q >> 1) & 7;
    const int half = q & 1;
    const float* gp = x + (((size_t)(b * C_ + c) * H_ + (h0 + rr)) * W_ + w0 + half * 4);
    unsigned loff = (unsigned)(uintptr_t)(&sx[c * 64 + rr * 8 + half * 4]);
    unsigned long long gaddr = (unsigned long long)(uintptr_t)gp;
    asm volatile("global_load_async_to_lds_b128 %0, %1, off"
                 :: "v"(loff), "v"(gaddr) : "memory");
  }
  asm volatile("s_wait_asynccnt 0" ::: "memory");
  __syncthreads();

  // ---- per-pixel LN stats over C (4 partial groups x 64 pixels) ----
  {
    const int p = tid & 63;
    const int g = tid >> 6;
    float s = 0.f, ss = 0.f;
    for (int c = g; c < C_; c += 4) {
      float v = sx[c * 64 + p];
      s += v; ss += v * v;
    }
    s_sum[g][p] = s;
    s_sq[g][p]  = ss;
  }
  __syncthreads();
  if (tid < 64) {
    float s  = s_sum[0][tid] + s_sum[1][tid] + s_sum[2][tid] + s_sum[3][tid];
    float ss = s_sq[0][tid]  + s_sq[1][tid]  + s_sq[2][tid]  + s_sq[3][tid];
    float m  = s * (1.0f / C_);
    float var = ss * (1.0f / C_) - m * m;
    float r  = rsqrtf(var + EPS_);
    s_mu[tid] = m;
    s_rstd[tid] = r;
    const int hh = h0 + (tid >> 3);
    const int ww = w0 + (tid & 7);
    const size_t pix = ((size_t)b * H_ + hh) * W_ + ww;
    mu_g[pix] = m;
    rstd_g[pix] = r;
  }
  __syncthreads();
  if (tid == 0) {
    float acc = 0.f;
    for (int q = 0; q < 64; ++q) acc += s_mu[q] * s_rstd[q];
    s_S = acc;
  }
  __syncthreads();

  // ---- per-channel patch mean of normalized values ----
  {
    const int c = tid;
    const int lane = tid & 31;
    float acc = 0.f;
    for (int q = 0; q < 64; ++q) {
      const int p = (q + lane) & 63;             // stagger: bank-conflict free
      acc += sx[c * 64 + p] * s_rstd[p];
    }
    const float pmv = gamma[c] * ((acc - s_S) * (1.0f / 64.0f)) + beta[c];
    pm_g[(size_t)pidx * C_ + c] = f2bf(pmv);
  }
}

// ---------------------------------------------------------------------------
// K2: gating MLP batched over all 8192 patches with bf16 WMMA.
//   h = silu(pm @ w1^T)   [8192x256]x[256x32]
//   g = sigmoid(h @ w2^T) [8192x32]x[32x256]
// One wave per 16-patch M-tile; 8 waves per block.
// ---------------------------------------------------------------------------
__global__ __launch_bounds__(256) void k2_mlp(
    const unsigned short* __restrict__ pm, const float* __restrict__ w1,
    const float* __restrict__ w2, float* __restrict__ gate)
{
  __shared__ unsigned short sh[8][16 * 32];   // per-wave silu(h) tile, bf16

  const int lane  = threadIdx.x & 31;
  const int wid   = threadIdx.x >> 5;
  const int mbase = (blockIdx.x * 8 + wid) * 16;
  const int half  = lane >> 4;
  const int l15   = lane & 15;

  union Frag { v16bf v; unsigned u[8]; };

  v8f acc0 = {0.f,0.f,0.f,0.f,0.f,0.f,0.f,0.f};
  v8f acc1 = {0.f,0.f,0.f,0.f,0.f,0.f,0.f,0.f};

#pragma unroll
  for (int ks = 0; ks < 8; ++ks) {
    // A fragment: 16-bit A layout (lane = M, halves/VGPRs = K)
    Frag a;
    const unsigned short* prow = pm + (size_t)(mbase + l15) * C_ + ks * 32 + half * 8;
#pragma unroll
    for (int v = 0; v < 8; ++v) {
      const int k0 = (v < 4) ? (2 * v) : (8 + 2 * v);   // pairs K=k0,k0+1
      a.u[v] = *(const unsigned*)(prow + k0);
    }
    // B fragments: lane = N, vector element = K (same order as A);  B[k][n] = w1[n][k]
    const int kb = ks * 32 + half * 8;
    Frag b0, b1;
#pragma unroll
    for (int v = 0; v < 8; ++v) {
      const int kk = kb + ((v < 4) ? (2 * v) : (8 + 2 * v));
      unsigned lo0 = f2bf(w1[l15 * C_ + kk]);
      unsigned hi0 = f2bf(w1[l15 * C_ + kk + 1]);
      b0.u[v] = lo0 | (hi0 << 16);
      unsigned lo1 = f2bf(w1[(16 + l15) * C_ + kk]);
      unsigned hi1 = f2bf(w1[(16 + l15) * C_ + kk + 1]);
      b1.u[v] = lo1 | (hi1 << 16);
    }
    acc0 = __builtin_amdgcn_wmma_f32_16x16x32_bf16(false, a.v, false, b0.v,
                                                   (short)0, acc0, false, false);
    acc1 = __builtin_amdgcn_wmma_f32_16x16x32_bf16(false, a.v, false, b1.v,
                                                   (short)0, acc1, false, false);
  }

  // SiLU, re-layout C/D -> A via per-wave LDS tile
#pragma unroll
  for (int j = 0; j < 8; ++j) {
    const int m = j + 8 * half;
    sh[wid][m * 32 + l15]      = f2bf(silu_(acc0[j]));
    sh[wid][m * 32 + 16 + l15] = f2bf(silu_(acc1[j]));
  }

  Frag a2;
#pragma unroll
  for (int v = 0; v < 8; ++v) {
    const int k0 = ((v < 4) ? (2 * v) : (8 + 2 * v)) + half * 8;
    const unsigned short* hp = &sh[wid][l15 * 32 + k0];
    a2.u[v] = ((unsigned)hp[0]) | (((unsigned)hp[1]) << 16);
  }

  const int kb2 = half * 8;   // K base for this lane half
  const v8f zero = {0.f,0.f,0.f,0.f,0.f,0.f,0.f,0.f};
  for (int nt = 0; nt < 16; ++nt) {
    Frag b2;
#pragma unroll
    for (int v = 0; v < 8; ++v) {
      const int kk = kb2 + ((v < 4) ? (2 * v) : (8 + 2 * v));
      unsigned lo = f2bf(w2[(nt * 16 + l15) * CR_ + kk]);
      unsigned hi = f2bf(w2[(nt * 16 + l15) * CR_ + kk + 1]);
      b2.u[v] = lo | (hi << 16);
    }
    v8f acc = __builtin_amdgcn_wmma_f32_16x16x32_bf16(false, a2.v, false, b2.v,
                                                      (short)0, zero, false, false);
#pragma unroll
    for (int j = 0; j < 8; ++j) {
      const int m = j + 8 * half;
      gate[(size_t)(mbase + m) * C_ + nt * 16 + l15] = sigm_(acc[j]);
    }
  }
}

// ---------------------------------------------------------------------------
// K3: streaming apply: out = (x - mu) * rstd * (gamma*g) + beta*g. 128-bit.
// Non-temporal on the 512 MiB streams; mu/rstd/gate stay cacheable (reused).
// ---------------------------------------------------------------------------
__global__ __launch_bounds__(256) void k3_apply(
    const v4f* __restrict__ x4, const float* __restrict__ gamma,
    const float* __restrict__ beta, const v4f* __restrict__ mu4,
    const v4f* __restrict__ rstd4, const float* __restrict__ gate,
    v4f* __restrict__ out4)
{
  const int idx = blockIdx.x * 256 + threadIdx.x;   // < 2^25
  const int w4 = idx & 63;              // W/4 = 64
  const int h  = (idx >> 6) & 255;
  const int c  = (idx >> 14) & 255;
  const int b  = idx >> 22;

  const v4f xv = __builtin_nontemporal_load(&x4[idx]);
  const int pix = (((b << 8) | h) << 6) | w4;       // [B,H,W/4]
  const v4f m = mu4[pix];
  const v4f r = rstd4[pix];
  const float g  = gate[((((b << 5) | (h >> 3)) << 5 | (w4 >> 1)) << 8) + c];
  const float ga = gamma[c] * g;
  const float be = beta[c] * g;

  v4f o;
#pragma unroll
  for (int e = 0; e < 4; ++e)
    o[e] = (xv[e] - m[e]) * r[e] * ga + be;
  __builtin_nontemporal_store(o, &out4[idx]);
}

// ---------------------------------------------------------------------------
extern "C" void kernel_launch(void* const* d_in, const int* in_sizes, int n_in,
                              void* d_out, int out_size, void* d_ws, size_t ws_size,
                              hipStream_t stream) {
  const float* x     = (const float*)d_in[0];
  const float* gamma = (const float*)d_in[1];
  const float* beta  = (const float*)d_in[2];
  const float* w1    = (const float*)d_in[3];
  const float* w2    = (const float*)d_in[4];
  float* out = (float*)d_out;

  char* ws = (char*)d_ws;
  float*          mu   = (float*)ws;                                   // 2 MiB
  float*          rstd = (float*)(ws + (size_t)2 * 1024 * 1024);       // 2 MiB
  unsigned short* pm   = (unsigned short*)(ws + (size_t)4 * 1024 * 1024); // 4 MiB
  float*          gate = (float*)(ws + (size_t)8 * 1024 * 1024);       // 8 MiB

  k1_patch_stats<<<NPATCH_, 256, 0, stream>>>(x, gamma, beta, mu, rstd, pm);
  k2_mlp<<<NPATCH_ / 128, 256, 0, stream>>>(pm, w1, w2, gate);
  k3_apply<<<(B_ * C_ * H_ * W_ / 4) / 256, 256, 0, stream>>>(
      (const v4f*)x, gamma, beta, (const v4f*)mu, (const v4f*)rstd,
      gate, (v4f*)out);
}
